// CrossAttenHead_6047313952934
// MI455X (gfx1250) — matrix operand order repaired and checked by reading, not compiled, not run on hardware
//
#include <hip/hip_runtime.h>
#include <hip/hip_bf16.h>

// ---------------------------------------------------------------------------
// CDNA5 (gfx1250) wave32 WMMA implementation of CrossAttenHead
// ---------------------------------------------------------------------------

typedef __attribute__((ext_vector_type(16))) _Float16 v16h;
typedef __attribute__((ext_vector_type(8)))  float    v8f;
typedef __attribute__((ext_vector_type(4)))  int      v4i;

#define HID    256
#define NPROP  100
#define NCLS   80
#define NHEADS 8
#define HWSZ   16384           // 128*128
#define BATCH  8

// ---------------- wave32 reductions ----------------
__device__ inline float wred_sum(float v) {
#pragma unroll
  for (int m = 16; m > 0; m >>= 1) v += __shfl_xor(v, m, 32);
  return v;
}

// ---------------------------------------------------------------------------
// Generic batched GEMM: C[b] = op(A[b] @ B[b] + bias)
//   A(m,k) = A[b*sA + m*ldaM + k]          (A always K-contiguous)
//   B(k,n) = BKMAJ ? B[b*sB + k + n*ldbN]  (K-contiguous, e.g. features as cols)
//                  : B[b*sB + k*ldbK + n]  (row-major)
// NT 16-wide N-subtiles per wave; one A fragment feeds NT v_wmma ops.
// OOB rows/cols: addresses clamped (results land in unstored C lanes).
// OOB K handled only in an explicit tail block (K%32, only the point-GEMM).
// ---------------------------------------------------------------------------
template <int MASK_A, int RELU, int NT, int BKMAJ, int VECA>
__global__ __launch_bounds__(256) void wmma_gemm_kernel(
    const float* __restrict__ A, long long sA, int ldaM,
    const float* __restrict__ B, long long sB, int ldbK, int ldbN,
    const float* __restrict__ bias,
    float* __restrict__ C, long long sC, int ldc,
    int M, int N, int K)
{
  const int lane   = threadIdx.x & 31;
  const int wave   = threadIdx.x >> 5;
  const int tilesM = (M + 15) >> 4;
  const int tilesN = (N + 15) >> 4;
  const int tilesG = (tilesN + NT - 1) / NT;
  const int tile   = blockIdx.x * 8 + wave;          // wave-uniform
  if (tile >= tilesM * tilesG) return;
  const int mt  = tile / tilesG;
  const int ntg = tile % tilesG;
  const int m0  = mt << 4;
  const int n0  = (ntg * NT) << 4;
  const int b   = blockIdx.z;

  const int row   = lane & 15;
  const int col   = lane & 15;
  const int khalf = (lane >> 4) & 1;

  // clamped base pointers (OOB rows/cols only feed unstored outputs)
  const int mA = min(m0 + row, M - 1);
  const float* pA = A + (long long)b * sA + (long long)mA * ldaM;

  const float* pB[NT];
#pragma unroll
  for (int i = 0; i < NT; ++i) {
    const int n = min(n0 + (i << 4) + col, N - 1);
    pB[i] = B + (long long)b * sB + (BKMAJ ? (long long)n * ldbN : (long long)n);
  }

  // fragment K-offsets (ISA 7.12.2 f16 layouts), loop-invariant
  int offA[8], offB[8];
#pragma unroll
  for (int j = 0; j < 8; ++j) {
    offA[j] = ((j >> 2) << 4) + (khalf << 3) + ((j & 3) << 1);
    offB[j] = ((j >> 2) << 4) + (khalf << 3) + ((j & 3) << 1);
  }

  v8f acc[NT];
#pragma unroll
  for (int i = 0; i < NT; ++i) acc[i] = (v8f){0,0,0,0,0,0,0,0};

  const int Kmain = K & ~31;
  for (int k0 = 0; k0 < Kmain; k0 += 32) {
    v16h a;
#pragma unroll
    for (int j = 0; j < 8; ++j) {
      float x, y;
      if (VECA) {
        const float2 v = *(const float2*)(pA + k0 + offA[j]);
        x = v.x; y = v.y;
      } else {
        x = pA[k0 + offA[j]];
        y = pA[k0 + offA[j] + 1];
      }
      if (MASK_A) { x = (x > 0.f) ? 1.f : 0.f; y = (y > 0.f) ? 1.f : 0.f; }
      a[2 * j]     = (_Float16)x;
      a[2 * j + 1] = (_Float16)y;
    }
    if (k0 + 32 < Kmain) __builtin_prefetch(pA + k0 + 32, 0, 1);
#pragma unroll
    for (int i = 0; i < NT; ++i) {
      v16h bf;
#pragma unroll
      for (int j = 0; j < 8; ++j) {
        if (BKMAJ) {
          const float2 v = *(const float2*)(pB[i] + k0 + offB[j]);
          bf[2 * j]     = (_Float16)v.x;
          bf[2 * j + 1] = (_Float16)v.y;
        } else {
          bf[2 * j]     = (_Float16)pB[i][(k0 + offB[j]) * ldbK];
          bf[2 * j + 1] = (_Float16)pB[i][(k0 + offB[j] + 1) * ldbK];
        }
      }
      acc[i] = __builtin_amdgcn_wmma_f32_16x16x32_f16(
          false, a, false, bf, (short)0, acc[i], false, false);
    }
  }

  if (K & 31) {  // K tail (only the point-GEMM, K=100): clamped + masked loads
    const int k0 = Kmain;
    v16h a;
#pragma unroll
    for (int j = 0; j < 8; ++j) {
#pragma unroll
      for (int t = 0; t < 2; ++t) {
        const int k = k0 + offA[j] + t;
        float x = pA[min(k, K - 1)];
        if (MASK_A) x = (x > 0.f) ? 1.f : 0.f;
        a[2 * j + t] = (_Float16)((k < K) ? x : 0.f);
      }
    }
#pragma unroll
    for (int i = 0; i < NT; ++i) {
      v16h bf;
#pragma unroll
      for (int j = 0; j < 8; ++j) {
#pragma unroll
        for (int t = 0; t < 2; ++t) {
          const int k  = k0 + offB[j] + t;
          const int kc = min(k, K - 1);
          const float x = BKMAJ ? pB[i][kc] : pB[i][kc * ldbK];
          bf[2 * j + t] = (_Float16)((k < K) ? x : 0.f);
        }
      }
      acc[i] = __builtin_amdgcn_wmma_f32_16x16x32_f16(
          false, a, false, bf, (short)0, acc[i], false, false);
    }
  }

  // epilogue: C layout -> VGPR r = (M = m0 + r + 8*khalf, N = n0 + i*16 + col)
#pragma unroll
  for (int i = 0; i < NT; ++i) {
    const int n = n0 + (i << 4) + col;
    if (n < N) {
      const float bv = bias ? bias[n] : 0.f;
#pragma unroll
      for (int r = 0; r < 8; ++r) {
        const int m = m0 + r + (khalf << 3);
        if (m < M) {
          float v = acc[i][r] + bv;
          if (RELU) v = fmaxf(v, 0.f);
          C[(long long)b * sC + (long long)m * ldc + n] = v;
        }
      }
    }
  }
}

// ---------------------------------------------------------------------------
// LayerNorm over last dim N (wave per row).  out = LN(x (+res)) * g + b, opt ReLU.
// ---------------------------------------------------------------------------
__global__ __launch_bounds__(256) void ln_kernel(
    const float* __restrict__ x, const float* __restrict__ res,
    const float* __restrict__ g, const float* __restrict__ bb,
    float* __restrict__ out, int rows, int N, int relu)
{
  const int lane = threadIdx.x & 31;
  const int row  = blockIdx.x * 8 + (threadIdx.x >> 5);
  if (row >= rows) return;
  const float* xr = x + (long long)row * N;
  const float* rr = res ? res + (long long)row * N : nullptr;
  float s = 0.f, s2 = 0.f;
  for (int i = lane; i < N; i += 32) {
    float v = xr[i] + (rr ? rr[i] : 0.f);
    s += v; s2 += v * v;
  }
  s = wred_sum(s); s2 = wred_sum(s2);
  const float mean = s / N;
  const float var  = s2 / N - mean * mean;
  const float rstd = rsqrtf(var + 1e-5f);
  float* orow = out + (long long)row * N;
  for (int i = lane; i < N; i += 32) {
    float v = xr[i] + (rr ? rr[i] : 0.f);
    v = (v - mean) * rstd * g[i] + bb[i];
    if (relu) v = fmaxf(v, 0.f);
    orow[i] = v;
  }
}

// ---------------------------------------------------------------------------
// dysep depthwise K=3 conv along channel dim + ReLU.
// ---------------------------------------------------------------------------
__global__ __launch_bounds__(256) void dw_conv_kernel(
    const float* __restrict__ w, int ldw,
    const float* __restrict__ val, float* __restrict__ depth, int rows)
{
  const int idx = blockIdx.x * blockDim.x + threadIdx.x;
  const int row = idx >> 8, c = idx & 255;
  if (row >= rows) return;
  const float* wr = w   + (long long)row * ldw;
  const float* vr = val + (long long)row * HID;
  float acc = wr[1] * vr[c];
  if (c > 0)       acc += wr[0] * vr[c - 1];
  if (c < HID - 1) acc += wr[2] * vr[c + 1];
  depth[(long long)row * HID + c] = fmaxf(acc, 0.f);
}

// ---------------------------------------------------------------------------
// MHA core: one wave per (b, head, q).  lane == head-dim index (d = 32 == wave32).
// ---------------------------------------------------------------------------
__global__ __launch_bounds__(256) void attn_kernel(
    const float* __restrict__ qkv, float* __restrict__ o)
{
  __shared__ float sc[8][NPROP];
  const int lane = threadIdx.x & 31;
  const int wave = threadIdx.x >> 5;
  const int wid  = blockIdx.x * 8 + wave;
  if (wid >= BATCH * NHEADS * NPROP) return;
  const int q = wid % NPROP;
  const int h = (wid / NPROP) % NHEADS;
  const int b = wid / (NPROP * NHEADS);

  const float* base = qkv + (long long)b * NPROP * 768;
  const float qv = base[(long long)q * 768 + h * 32 + lane];
  const float scale = 0.17677669529663687f;  // 1/sqrt(32)

  float mx = -1e30f;
  for (int k = 0; k < NPROP; ++k) {
    float t = qv * base[(long long)k * 768 + 256 + h * 32 + lane];
    t = wred_sum(t) * scale;        // same value on all lanes
    if (lane == 0) sc[wave][k] = t;
    mx = fmaxf(mx, t);
  }
  float ssum = 0.f;
  for (int k = lane; k < NPROP; k += 32) {
    const float p = __expf(sc[wave][k] - mx);
    sc[wave][k] = p;
    ssum += p;
  }
  ssum = wred_sum(ssum);
  float od = 0.f;
  for (int k = 0; k < NPROP; ++k)
    od += sc[wave][k] * base[(long long)k * 768 + 512 + h * 32 + lane];
  o[((long long)b * NPROP + q) * HID + h * 32 + lane] = od / ssum;
}

// ---------------------------------------------------------------------------
// Bulk copy via the CDNA5 async LDS pipeline when available:
// GLOBAL_LOAD_ASYNC_TO_LDS_B128 -> s_wait_asynccnt -> GLOBAL_STORE_ASYNC_FROM_LDS_B128
// (ASYNCcnt-tracked, bypasses VGPRs).  Guarded with __has_builtin; falls back
// to a plain float4 copy so the build can never regress.
// Builtin signature (from clang diagnostics): (addrspace(1) v4i*,
// addrspace(3) v4i*, imm offset, imm cpol).
// ---------------------------------------------------------------------------
#if __has_builtin(__builtin_amdgcn_global_load_async_to_lds_b128) && \
    __has_builtin(__builtin_amdgcn_global_store_async_from_lds_b128)
#define HAVE_ASYNC_LDS 1
#else
#define HAVE_ASYNC_LDS 0
#endif

__device__ inline void wait_asynccnt0() {
#if __has_builtin(__builtin_amdgcn_s_wait_asynccnt)
  __builtin_amdgcn_s_wait_asynccnt(0);
#else
  asm volatile("s_wait_asynccnt 0x0" ::: "memory");
#endif
}

__global__ __launch_bounds__(256) void copy_kernel(const float* __restrict__ src,
                                                   float* __restrict__ dst, long long n4)
{
  const long long i = (long long)blockIdx.x * blockDim.x + threadIdx.x;
  if (i >= n4) return;
#if HAVE_ASYNC_LDS
  __shared__ float4 stage[256];
  __builtin_amdgcn_global_load_async_to_lds_b128(
      (__attribute__((address_space(1))) v4i*)(src + i * 4),
      (__attribute__((address_space(3))) v4i*)&stage[threadIdx.x], 0, 0);
  wait_asynccnt0();
  __builtin_amdgcn_global_store_async_from_lds_b128(
      (__attribute__((address_space(1))) v4i*)(dst + i * 4),
      (__attribute__((address_space(3))) v4i*)&stage[threadIdx.x], 0, 0);
  wait_asynccnt0();
#else
  ((float4*)dst)[i] = ((const float4*)src)[i];
#endif
}

// ---------------------------------------------------------------------------
// Host-side orchestration
// ---------------------------------------------------------------------------
template <int MASK_A, int RELU, int NT, int BKMAJ, int VECA>
static inline void launch_gemm(const float* A, long long sA, int ldaM,
                               const float* B, long long sB, int ldbK, int ldbN,
                               const float* bias, float* C, long long sC, int ldc,
                               int M, int N, int K, hipStream_t s)
{
  const int tilesM = (M + 15) / 16, tilesN = (N + 15) / 16;
  const int tiles  = tilesM * ((tilesN + NT - 1) / NT);
  dim3 grid((tiles + 7) / 8, 1, BATCH);
  wmma_gemm_kernel<MASK_A, RELU, NT, BKMAJ, VECA>
      <<<grid, 256, 0, s>>>(A, sA, ldaM, B, sB, ldbK, ldbN, bias, C, sC, ldc, M, N, K);
}

static inline void layernorm(const float* x, const float* res, const float* g, const float* b,
                             float* out, int rows, int N, int relu, hipStream_t s)
{
  ln_kernel<<<(rows + 7) / 8, 256, 0, s>>>(x, res, g, b, out, rows, N, relu);
}

// workspace offsets (floats)
#define OFF_F     0LL
#define OFF_W     204800LL                 // B*100*112
#define OFF_TMP   (OFF_W   + 89600LL)
#define OFF_DEPTH (OFF_TMP + 204800LL)
#define OFF_F2    (OFF_DEPTH + 204800LL)
#define OFF_KK2   (OFF_F2  + 204800LL)
#define OFF_QKV   (OFF_KK2 + 204800LL)     // B*100*768
#define OFF_O     (OFF_QKV + 614400LL)
#define OFF_MO    (OFF_O   + 204800LL)
#define OFF_KK3   (OFF_MO  + 204800LL)
#define OFF_H     (OFF_KK3 + 204800LL)     // B*100*2048
#define OFF_F3    (OFF_H   + 1638400LL)
#define OFF_OBJ   (OFF_F3  + 204800LL)
#define OFF_CG    (OFF_OBJ + 204800LL)
#define OFF_CF    (OFF_CG  + 204800LL)
#define OFF_MG    (OFF_CF  + 204800LL)
#define OFF_MF    (OFF_MG  + 204800LL)
#define OFF_MK    (OFF_MF  + 204800LL)

// output offsets (floats)
#define OUT_CLS   0LL                       // 8*100*81    = 64800
#define OUT_MASK  64800LL                   // 8*100*16384 = 13107200
#define OUT_OBJ   (64800LL + 13107200LL)    // 8*100*256   = 204800

extern "C" void kernel_launch(void* const* d_in, const int* in_sizes, int n_in,
                              void* d_out, int out_size, void* d_ws, size_t ws_size,
                              hipStream_t stream)
{
  const float* features  = (const float*)d_in[0];
  const float* prop_k    = (const float*)d_in[1];
  const float* mask_pred = (const float*)d_in[2];
  const float* fW = (const float*)d_in[3];  const float* fb = (const float*)d_in[4];
  const float* fng = (const float*)d_in[5]; const float* fnb = (const float*)d_in[6];
  const float* f_norm_g = (const float*)d_in[7]; const float* f_norm_b = (const float*)d_in[8];
  const float* kW = (const float*)d_in[9];  const float* kb = (const float*)d_in[10];
  const float* kng = (const float*)d_in[11]; const float* knb = (const float*)d_in[12];
  const float* k_norm_g = (const float*)d_in[13]; const float* k_norm_b = (const float*)d_in[14];
  const float* attn_in_w = (const float*)d_in[15]; const float* attn_in_b = (const float*)d_in[16];
  const float* attn_out_w = (const float*)d_in[17]; const float* attn_out_b = (const float*)d_in[18];
  const float* s_norm_g = (const float*)d_in[19]; const float* s_norm_b = (const float*)d_in[20];
  const float* ffn_w1 = (const float*)d_in[21]; const float* ffn_b1 = (const float*)d_in[22];
  const float* ffn_w2 = (const float*)d_in[23]; const float* ffn_b2 = (const float*)d_in[24];
  const float* ffn_norm_g = (const float*)d_in[25]; const float* ffn_norm_b = (const float*)d_in[26];
  const float* cls_w = (const float*)d_in[27];
  const float* cls_ln_g = (const float*)d_in[28]; const float* cls_ln_b = (const float*)d_in[29];
  const float* fc_cls_w = (const float*)d_in[30]; const float* fc_cls_b = (const float*)d_in[31];
  const float* mask_w = (const float*)d_in[32];
  const float* mask_ln_g = (const float*)d_in[33]; const float* mask_ln_b = (const float*)d_in[34];
  const float* fc_mask_w = (const float*)d_in[35]; const float* fc_mask_b = (const float*)d_in[36];

  float* ws  = (float*)d_ws;
  float* out = (float*)d_out;
  const int ROWS = BATCH * NPROP;  // 800

  // 1) pooling einsum: f[b,n,c] = sum_hw hard(mask)[b,n,hw] * feat[b,c,hw]
  //    B is K(hw)-contiguous -> BKMAJ, float2 loads on both operands.
  launch_gemm<1, 0, 2, 1, 1>(mask_pred, (long long)NPROP * HWSZ, HWSZ,
                             features, (long long)HID * HWSZ, 1, HWSZ,
                             nullptr, ws + OFF_F, (long long)NPROP * HID, HID,
                             NPROP, HID, HWSZ, stream);

  // 2) dysep #1 (query=prop_k, value=f)
  launch_gemm<0, 0, 2, 0, 1>(prop_k, (long long)NPROP * HID, HID,
                             fW, 0, NPROP + 3, 1, fb,
                             ws + OFF_W, (long long)NPROP * 112, 112,
                             NPROP, NPROP + 3, HID, stream);
  dw_conv_kernel<<<ROWS, 256, 0, stream>>>(ws + OFF_W, 112, ws + OFF_F, ws + OFF_DEPTH, ROWS);
  launch_gemm<0, 0, 4, 0, 0>(ws + OFF_W + 3, (long long)NPROP * 112, 112,   // pw (+3 -> unaligned A)
                             ws + OFF_DEPTH, (long long)NPROP * HID, HID, 1, nullptr,
                             ws + OFF_TMP, (long long)NPROP * HID, HID,
                             NPROP, HID, NPROP, stream);
  layernorm(ws + OFF_TMP, nullptr, fng, fnb, ws + OFF_TMP, ROWS, HID, 0, stream);
  layernorm(ws + OFF_TMP, ws + OFF_F, f_norm_g, f_norm_b, ws + OFF_F2, ROWS, HID, 0, stream);

  // 3) dysep #2 (query=prop_k, value=f2)
  launch_gemm<0, 0, 2, 0, 1>(prop_k, (long long)NPROP * HID, HID,
                             kW, 0, NPROP + 3, 1, kb,
                             ws + OFF_W, (long long)NPROP * 112, 112,
                             NPROP, NPROP + 3, HID, stream);
  dw_conv_kernel<<<ROWS, 256, 0, stream>>>(ws + OFF_W, 112, ws + OFF_F2, ws + OFF_DEPTH, ROWS);
  launch_gemm<0, 0, 4, 0, 0>(ws + OFF_W + 3, (long long)NPROP * 112, 112,
                             ws + OFF_DEPTH, (long long)NPROP * HID, HID, 1, nullptr,
                             ws + OFF_TMP, (long long)NPROP * HID, HID,
                             NPROP, HID, NPROP, stream);
  layernorm(ws + OFF_TMP, nullptr, kng, knb, ws + OFF_TMP, ROWS, HID, 0, stream);
  layernorm(ws + OFF_TMP, ws + OFF_F2, k_norm_g, k_norm_b, ws + OFF_KK2, ROWS, HID, 0, stream);

  // 4) MHA
  launch_gemm<0, 0, 4, 0, 1>(ws + OFF_KK2, (long long)NPROP * HID, HID,
                             attn_in_w, 0, 3 * HID, 1, attn_in_b,
                             ws + OFF_QKV, (long long)NPROP * 768, 768,
                             NPROP, 3 * HID, HID, stream);
  attn_kernel<<<(BATCH * NHEADS * NPROP) / 8, 256, 0, stream>>>(ws + OFF_QKV, ws + OFF_O);
  launch_gemm<0, 0, 4, 0, 1>(ws + OFF_O, (long long)NPROP * HID, HID,
                             attn_out_w, 0, HID, 1, attn_out_b,
                             ws + OFF_MO, (long long)NPROP * HID, HID,
                             NPROP, HID, HID, stream);
  layernorm(ws + OFF_MO, ws + OFF_KK2, s_norm_g, s_norm_b, ws + OFF_KK3, ROWS, HID, 0, stream);

  // 5) FFN
  launch_gemm<0, 1, 4, 0, 1>(ws + OFF_KK3, (long long)NPROP * HID, HID,
                             ffn_w1, 0, 2048, 1, ffn_b1,
                             ws + OFF_H, (long long)NPROP * 2048, 2048,
                             NPROP, 2048, HID, stream);
  launch_gemm<0, 0, 4, 0, 1>(ws + OFF_H, (long long)NPROP * 2048, 2048,
                             ffn_w2, 0, HID, 1, ffn_b2,
                             ws + OFF_F3, (long long)NPROP * HID, HID,
                             NPROP, HID, 2048, stream);
  layernorm(ws + OFF_F3, ws + OFF_KK3, ffn_norm_g, ffn_norm_b, ws + OFF_OBJ, ROWS, HID, 0, stream);

  // 6) cls head (train_flag==1 path)
  launch_gemm<0, 0, 4, 0, 1>(ws + OFF_OBJ, (long long)NPROP * HID, HID,
                             cls_w, 0, HID, 1, nullptr,
                             ws + OFF_CG, (long long)NPROP * HID, HID,
                             NPROP, HID, HID, stream);
  layernorm(ws + OFF_CG, nullptr, cls_ln_g, cls_ln_b, ws + OFF_CF, ROWS, HID, 1, stream);
  launch_gemm<0, 0, 2, 0, 1>(ws + OFF_CF, (long long)NPROP * HID, HID,
                             fc_cls_w, 0, NCLS + 1, 1, fc_cls_b,
                             out + OUT_CLS, (long long)NPROP * (NCLS + 1), NCLS + 1,
                             NPROP, NCLS + 1, HID, stream);

  // 7) mask head: 3x (linear -> LN -> relu), then fc_mask
  const float* msrc = ws + OFF_OBJ;
  for (int i = 0; i < 3; ++i) {
    launch_gemm<0, 0, 4, 0, 1>(msrc, (long long)NPROP * HID, HID,
                               mask_w + (long long)i * HID * HID, 0, HID, 1, nullptr,
                               ws + OFF_MG, (long long)NPROP * HID, HID,
                               NPROP, HID, HID, stream);
    layernorm(ws + OFF_MG, nullptr, mask_ln_g + i * HID, mask_ln_b + i * HID,
              ws + OFF_MF, ROWS, HID, 1, stream);
    msrc = ws + OFF_MF;
  }
  launch_gemm<0, 0, 4, 0, 1>(ws + OFF_MF, (long long)NPROP * HID, HID,
                             fc_mask_w, 0, HID, 1, fc_mask_b,
                             ws + OFF_MK, (long long)NPROP * HID, HID,
                             NPROP, HID, HID, stream);

  // 8) obj_out = obj (K2==1 so transpose/reshape is identity) — async LDS bounce
  copy_kernel<<<(204800 / 4 + 255) / 256, 256, 0, stream>>>(ws + OFF_OBJ, out + OUT_OBJ,
                                                            204800LL / 4);

  // 9) expansion einsum: new_mask_preds[b,q,hw] = mk[b,q,:] @ feat[b,:,hw]
  //    B = features row-major (ldbK = HWSZ), N = 16384 streamed wide (NT=4).
  launch_gemm<0, 0, 4, 0, 1>(ws + OFF_MK, (long long)NPROP * HID, HID,
                             features, (long long)HID * HWSZ, HWSZ, 1, nullptr,
                             out + OUT_MASK, (long long)NPROP * HWSZ, HWSZ,
                             NPROP, HWSZ, HID, stream);
}
